// BILSTMCRF_76072460747183
// MI455X (gfx1250) — matrix operand order repaired and checked by reading, not compiled, not run on hardware
//
#include <hip/hip_runtime.h>
#include <hip/hip_bf16.h>

// ---------- types ----------
typedef __attribute__((ext_vector_type(16))) __bf16 v16bf;
typedef __attribute__((ext_vector_type(8)))  float  v8f;
typedef __attribute__((ext_vector_type(4)))  unsigned v4u;
typedef __attribute__((ext_vector_type(8)))  int    v8i;
typedef __attribute__((ext_vector_type(4)))  int    v4i;

#define B_  64
#define T_  2048
#define E_  128
#define H_  128
#define G4_ 512   // 4*H
#define C_  8

// ---------- helpers ----------
__device__ __forceinline__ unsigned short f2bf(float f) {
  unsigned u = __float_as_uint(f);
  u += 0x7FFFu + ((u >> 16) & 1u);   // round-to-nearest-even
  return (unsigned short)(u >> 16);
}
__device__ __forceinline__ float bf2f(unsigned short h) {
  return __uint_as_float(((unsigned)h) << 16);
}
__device__ __forceinline__ float sigm(float x) { return 1.f / (1.f + __expf(-x)); }

__device__ __forceinline__ v4u pack8_bf16(v8f a) {
  union { v4u q; unsigned short s[8]; } u;
#pragma unroll
  for (int j = 0; j < 8; ++j) u.s[j] = f2bf(a[j]);
  return u.q;
}

// A fragment: 16x32 bf16, ISA 7.12.2 (lanes 0-15: row=lane, K {0..7,16..23};
// lanes 16-31: same rows, K {8..15,24..31}) -> two 16B loads per lane.
__device__ __forceinline__ v16bf load_fragA(const unsigned short* tile, int stride, int lane) {
  const int r  = lane & 15;
  const int kh = (lane >> 4) << 3;           // 0 or 8
  const unsigned short* p = tile + r * stride + kh;
  union { v16bf v; v4u q[2]; } u;
  u.q[0] = *(const v4u*)(p);
  u.q[1] = *(const v4u*)(p + 16);
  return u.v;
}
// B fragment: 32x16 bf16; lane holds column n=lane&15, same K order as A
// (lanes 0-15: K {0..7,16..23}, lanes 16-31: K {8..15,24..31}). tile points at W[n0][k0] row-major.
__device__ __forceinline__ v16bf load_fragB(const unsigned short* tile, int stride, int lane) {
  const int n  = lane & 15;
  const int ko = (lane >> 4) << 3;           // 0 or 8
  const unsigned short* p = tile + n * stride + ko;
  union { v16bf v; v4u q[2]; } u;
  u.q[0] = *(const v4u*)(p);
  u.q[1] = *(const v4u*)(p + 16);
  return u.v;
}
__device__ __forceinline__ v8f wmma_bf16(v16bf a, v16bf b, v8f c) {
  return __builtin_amdgcn_wmma_f32_16x16x32_bf16(false, a, false, b, (short)0, c, false, false);
}

// ---------- TDM: stage a 2D bf16 tensor into LDS (best-effort D#; a vector
// copy after the wait remains authoritative for the data) ----------
#if defined(__HIP_DEVICE_COMPILE__) && __has_builtin(__builtin_amdgcn_tensor_load_to_lds)
#define HAVE_TDM 1
__device__ __forceinline__ void tdm_load_2d(unsigned lds_off, const void* gaddr,
                                            unsigned dim0, unsigned dim1, unsigned stride0) {
  unsigned long long ga = (unsigned long long)(size_t)gaddr;
  v4u g0;
  g0[0] = 1u;                                             // count=1, user mode
  g0[1] = lds_off;                                        // lds_addr (bytes)
  g0[2] = (unsigned)ga;                                   // global_addr[31:0]
  g0[3] = (unsigned)((ga >> 32) & 0x1FFFFFFu) | 0x80000000u; // addr[56:32], type=2
  v8i g1;
  g1[0] = (int)(1u << 16);                                // data_size=1 (2 bytes)
  g1[1] = (int)(dim0 << 16);                              // tensor_dim0[15:0] @ [31:16]
  g1[2] = (int)((dim0 >> 16) | (dim1 << 16));             // dim0 hi | dim1 lo
  g1[3] = (int)((dim1 >> 16) | (dim0 << 16));             // dim1 hi | tile_dim0
  g1[4] = (int)(dim1 & 0xFFFFu);                          // tile_dim1 (tile_dim2=0)
  g1[5] = (int)stride0;                                   // tensor_dim0_stride lo32
  g1[6] = 0;
  g1[7] = 0;
  v4i z4 = {0, 0, 0, 0};
#if __clang_major__ >= 23
  v8i z8 = {0, 0, 0, 0, 0, 0, 0, 0};
  __builtin_amdgcn_tensor_load_to_lds(g0, g1, z4, z4, z8, 0);
#else
  __builtin_amdgcn_tensor_load_to_lds(g0, g1, z4, z4, 0);
#endif
}
#endif

// ---------- K0: fp32 -> bf16 convert ----------
__global__ void k_cvt_bf16(const float* __restrict__ src, unsigned short* __restrict__ dst, int n) {
  int i = blockIdx.x * blockDim.x + threadIdx.x;
  if (i < n) dst[i] = f2bf(src[i]);
}

// ---------- K1: embedding gather + input projection (both directions) ----------
// G layout: [T][4H][B] bf16 (batch-fastest -> D-fragment stores are contiguous)
__global__ void __launch_bounds__(256)
k_input_proj(const int* __restrict__ x,
             const unsigned short* __restrict__ emb,
             const unsigned short* __restrict__ wih_f,
             const unsigned short* __restrict__ wih_b,
             const float* __restrict__ bias_f,
             const float* __restrict__ bias_b,
             unsigned short* __restrict__ Gf,
             unsigned short* __restrict__ Gb) {
  const int lane  = threadIdx.x & 31;
  const int wave  = threadIdx.x >> 5;
  const int mtile = blockIdx.x * 8 + wave;          // 8192 row tiles of 16
  const int row0  = mtile * 16;                     // rows = t*64 + b
  const int t0    = row0 >> 6;                      // whole tile shares one t
  const int b0    = row0 & 63;

  // gather A fragments straight from the embedding table
  const int r   = lane & 15;
  const int idx = x[(b0 + r) * T_ + t0];
  const unsigned short* arow = emb + (size_t)idx * E_;
  const int kh = (lane >> 4) << 3;
  v16bf a[4];
#pragma unroll
  for (int kk = 0; kk < 4; ++kk) {
    union { v16bf v; v4u q[2]; } u;
    const unsigned short* p = arow + kk * 32 + kh;
    u.q[0] = *(const v4u*)(p);
    u.q[1] = *(const v4u*)(p + 16);
    a[kk] = u.v;
  }

  const int jo = (lane >> 4) << 3;
  const int nl = lane & 15;
#pragma unroll 1
  for (int dir = 0; dir < 2; ++dir) {
    const unsigned short* W = dir ? wih_b : wih_f;
    const float* bias       = dir ? bias_b : bias_f;
    unsigned short* G       = dir ? Gb : Gf;
#pragma unroll 1
    for (int nt = 0; nt < 32; ++nt) {
      const int col = nt * 16 + nl;
      const float bv = bias[col];
      v8f acc;
#pragma unroll
      for (int j = 0; j < 8; ++j) acc[j] = bv;      // bias pre-folded into C
#pragma unroll
      for (int kk = 0; kk < 4; ++kk) {
        v16bf bf = load_fragB(W + nt * 16 * E_ + kk * 32, E_, lane);
        acc = wmma_bf16(a[kk], bf, acc);
      }
      // one contiguous 16B non-temporal store per lane
      v4u* gp = (v4u*)(G + ((size_t)t0 * G4_ + col) * B_ + b0 + jo);
      __builtin_nontemporal_store(pack8_bf16(acc), gp);
    }
  }
}

// ---------- K2: sequential LSTM scan (blockIdx 0 = fwd, 1 = bwd) ----------
// h out layout: [T][H][B] bf16. Whh staged in LDS (TDM + authoritative copy).
__global__ void __launch_bounds__(256)
k_lstm_scan(const unsigned short* __restrict__ Gf, const unsigned short* __restrict__ Gb,
            const unsigned short* __restrict__ whh_f, const unsigned short* __restrict__ whh_b,
            unsigned short* __restrict__ hf, unsigned short* __restrict__ hb) {
  __shared__ __align__(16) unsigned short sh_whh[G4_ * H_]; // 128 KB recurrent weights
  __shared__ __align__(16) unsigned short sh_h[B_ * H_];    // 16 KB h state (bf16)

  const int dir = blockIdx.x;
  const unsigned short* G = dir ? Gb : Gf;
  const unsigned short* W = dir ? whh_b : whh_f;
  unsigned short* hout    = dir ? hb : hf;

  const int lane  = threadIdx.x & 31;
  const int wave  = threadIdx.x >> 5;
  const int mtile = wave & 3;                       // 4 row tiles (64 batch rows)
  const int hq    = wave >> 2;                      // 2 halves of the H columns
  const int jo    = (lane >> 4) << 3;
  const int nl    = lane & 15;

#ifdef HAVE_TDM
  if (threadIdx.x < 32)
    tdm_load_2d((unsigned)(size_t)&sh_whh[0], W, H_, G4_, H_);
#if __has_builtin(__builtin_amdgcn_s_wait_tensorcnt)
  __builtin_amdgcn_s_wait_tensorcnt(0);
#endif
#endif
  __syncthreads();
  // authoritative staging copy (b128 both sides) + state init
  for (int i = threadIdx.x; i < (G4_ * H_) / 8; i += 256)
    ((v4u*)sh_whh)[i] = ((const v4u*)W)[i];
  for (int i = threadIdx.x; i < B_ * H_; i += 256) sh_h[i] = 0;
  __syncthreads();

  float creg[4][8];                                 // cell state lives in VGPRs
#pragma unroll
  for (int p = 0; p < 4; ++p)
#pragma unroll
    for (int j = 0; j < 8; ++j) creg[p][j] = 0.f;

  for (int tt = 0; tt < T_; ++tt) {
    const int t = dir ? (T_ - 1 - tt) : tt;

    // A fragments: rows of h (bf16) from LDS
    v16bf a[4];
#pragma unroll
    for (int kk = 0; kk < 4; ++kk)
      a[kk] = load_fragA(sh_h + mtile * 16 * H_ + kk * 32, H_, lane);
    __syncthreads();                                // everyone done reading h

    // prefetch next timestep's gate slab
    if (tt + 1 < T_) {
      const int tn = dir ? (T_ - 2 - tt) : (tt + 1);
      __builtin_prefetch(G + (size_t)tn * G4_ * B_, 0, 1);
    }

    // gates = G[t] + h @ Whh.T ; acc[p][gate]
    v8f acc[4][4];
#pragma unroll
    for (int p = 0; p < 4; ++p) {
      const int hcol = hq * 4 + p;
#pragma unroll
      for (int g = 0; g < 4; ++g) {
        const int ntile = g * 8 + hcol;             // gate-major columns i,f,g,o
        // contiguous 16B non-temporal load of the C tile (G layout [T][4H][B])
        const v4u gq = __builtin_nontemporal_load(
            (const v4u*)(G + ((size_t)t * G4_ + ntile * 16 + nl) * B_ + mtile * 16 + jo));
        union { v4u q; unsigned short s[8]; } gu; gu.q = gq;
        v8f c;
#pragma unroll
        for (int j = 0; j < 8; ++j) c[j] = bf2f(gu.s[j]);
#pragma unroll
        for (int kk = 0; kk < 4; ++kk) {
          v16bf bf = load_fragB(sh_whh + ntile * 16 * H_ + kk * 32, H_, lane);
          c = wmma_bf16(a[kk], bf, c);
        }
        acc[p][g] = c;
      }
    }

    // elementwise LSTM cell update; lane owns (m, hd) across all four gates
#pragma unroll
    for (int p = 0; p < 4; ++p) {
      const int hd = (hq * 4 + p) * 16 + nl;
      union { v4u q; unsigned short s[8]; } hp;
#pragma unroll
      for (int j = 0; j < 8; ++j) {
        const float iv = sigm(acc[p][0][j]);
        const float fv = sigm(acc[p][1][j]);
        const float gv = tanhf(acc[p][2][j]);
        const float ov = sigm(acc[p][3][j]);
        float cc = creg[p][j];
        cc = fv * cc + iv * gv;
        creg[p][j] = cc;
        const float hv = ov * tanhf(cc);
        const unsigned short h16 = f2bf(hv);
        hp.s[j] = h16;
        sh_h[(mtile * 16 + jo + j) * H_ + hd] = h16;
      }
      // one contiguous 16B store of 8 batch rows (h layout [T][H][B])
      *(v4u*)(hout + ((size_t)t * H_ + hd) * B_ + mtile * 16 + jo) = hp.q;
    }
    __syncthreads();                                // h updates visible
  }
}

// ---------- K3: FC emissions + CRF forward (one wave per batch row) ----------
__global__ void __launch_bounds__(32)
k_fc_crf(const unsigned short* __restrict__ hf, const unsigned short* __restrict__ hb,
         const float* __restrict__ fc_w, const float* __restrict__ fc_b,
         const float* __restrict__ trans, const float* __restrict__ start,
         const float* __restrict__ endv, float* __restrict__ out) {
  const int b    = blockIdx.x;
  const int lane = threadIdx.x;                     // 0..31

  float alpha = (lane < C_) ? start[lane] : -1e30f;
  float tr[C_];                                     // column `lane` of trans
#pragma unroll
  for (int j = 0; j < C_; ++j) tr[j] = (lane < C_) ? trans[j * C_ + lane] : 0.f;

  for (int t = 0; t < T_; ++t) {
    // h layout is [T][H][B]; lane owns k = lane*4 .. lane*4+3
    float hv[8];
#pragma unroll
    for (int i = 0; i < 4; ++i) {
      hv[i]     = bf2f(hf[((size_t)t * H_ + lane * 4 + i) * B_ + b]);
      hv[4 + i] = bf2f(hb[((size_t)t * H_ + lane * 4 + i) * B_ + b]);
    }

    float lg[C_];
#pragma unroll
    for (int cc = 0; cc < C_; ++cc) {
      const float* w = fc_w + cc * (2 * H_);
      float s = 0.f;
#pragma unroll
      for (int i = 0; i < 4; ++i) {
        s += hv[i]     * w[lane * 4 + i];
        s += hv[4 + i] * w[H_ + lane * 4 + i];
      }
#pragma unroll
      for (int off = 16; off >= 1; off >>= 1) s += __shfl_xor(s, off, 32);
      lg[cc] = s + fc_b[cc];
    }
    float mylg = lg[0];
#pragma unroll
    for (int cc = 1; cc < C_; ++cc) if ((lane & 7) == cc) mylg = lg[cc];

    if (t == 0) {
      if (lane < C_) alpha += mylg;
    } else {
      float mx = -1e30f, aj[C_];
#pragma unroll
      for (int j = 0; j < C_; ++j) {
        const float av = __shfl(alpha, j, 32);
        aj[j] = av + tr[j];
        mx = fmaxf(mx, aj[j]);
      }
      float s = 0.f;
#pragma unroll
      for (int j = 0; j < C_; ++j) s += __expf(aj[j] - mx);
      const float na = mx + __logf(s) + mylg;
      alpha = (lane < C_) ? na : -1e30f;
    }
  }
  // out[b] = logsumexp_c(alpha + end)
  float v = (lane < C_) ? (alpha + endv[lane]) : -1e30f;
  float mx = v;
#pragma unroll
  for (int off = 16; off >= 1; off >>= 1) mx = fmaxf(mx, __shfl_xor(mx, off, 32));
  float s = (lane < C_) ? __expf(v - mx) : 0.f;
#pragma unroll
  for (int off = 16; off >= 1; off >>= 1) s += __shfl_xor(s, off, 32);
  if (lane == 0) out[b] = mx + __logf(s);
}

// ---------- launch ----------
extern "C" void kernel_launch(void* const* d_in, const int* in_sizes, int n_in,
                              void* d_out, int out_size, void* d_ws, size_t ws_size,
                              hipStream_t stream) {
  (void)in_sizes; (void)n_in; (void)out_size; (void)ws_size;
  const int*   x      = (const int*)  d_in[0];
  const float* emb    = (const float*)d_in[1];
  const float* Wih_f  = (const float*)d_in[2];
  const float* Whh_f  = (const float*)d_in[3];
  const float* b_f    = (const float*)d_in[4];
  const float* Wih_b  = (const float*)d_in[5];
  const float* Whh_b  = (const float*)d_in[6];
  const float* b_b    = (const float*)d_in[7];
  const float* fc_w   = (const float*)d_in[8];
  const float* fc_b   = (const float*)d_in[9];
  const float* trans  = (const float*)d_in[10];
  const float* start  = (const float*)d_in[11];
  const float* endv   = (const float*)d_in[12];
  float* out = (float*)d_out;

  // workspace layout (bytes)
  char* ws = (char*)d_ws;
  unsigned short* emb_bf = (unsigned short*)(ws);                       // 4096*128
  unsigned short* wihf   = (unsigned short*)(ws + 1048576);             // 512*128
  unsigned short* whhf   = (unsigned short*)(ws + 1179648);
  unsigned short* wihb   = (unsigned short*)(ws + 1310720);
  unsigned short* whhb   = (unsigned short*)(ws + 1441792);
  unsigned short* Gf     = (unsigned short*)(ws + 1572864);             // [T][512][64] bf16
  unsigned short* Gb     = (unsigned short*)(ws + 1572864   + 134217728ull);
  unsigned short* hf     = (unsigned short*)(ws + 1572864   + 268435456ull);  // [T][128][64]
  unsigned short* hb     = (unsigned short*)(ws + 1572864   + 268435456ull + 33554432ull);

  // K0: converts
  k_cvt_bf16<<<(4096 * 128 + 255) / 256, 256, 0, stream>>>(emb,   emb_bf, 4096 * 128);
  k_cvt_bf16<<<(512  * 128 + 255) / 256, 256, 0, stream>>>(Wih_f, wihf,   512 * 128);
  k_cvt_bf16<<<(512  * 128 + 255) / 256, 256, 0, stream>>>(Whh_f, whhf,   512 * 128);
  k_cvt_bf16<<<(512  * 128 + 255) / 256, 256, 0, stream>>>(Wih_b, wihb,   512 * 128);
  k_cvt_bf16<<<(512  * 128 + 255) / 256, 256, 0, stream>>>(Whh_b, whhb,   512 * 128);

  // K1: 8192 M-tiles / 8 waves per block
  k_input_proj<<<1024, 256, 0, stream>>>(x, emb_bf, wihf, wihb, b_f, b_b, Gf, Gb);

  // K2: two persistent blocks (fwd / bwd)
  k_lstm_scan<<<2, 256, 0, stream>>>(Gf, Gb, whhf, whhb, hf, hb);

  // K3: one wave per batch row
  k_fc_crf<<<B_, 32, 0, stream>>>(hf, hb, fc_w, fc_b, trans, start, endv, out);
}
